// MultiheadSelfAttention_28157805593363
// MI455X (gfx1250) — compile-verified, matrix operand order checked by reading, not run on hardware
//
#include <hip/hip_runtime.h>

// ---------------------------------------------------------------------------
// Types
// ---------------------------------------------------------------------------
typedef unsigned short u16;
typedef __attribute__((ext_vector_type(8)))  float   v8f;
typedef __attribute__((ext_vector_type(8)))  __bf16  v8bf;
typedef __attribute__((ext_vector_type(16))) __bf16  v16bf;

#define BB 4
#define TT 1024
#define DM 1024
#define NH 16
#define DH 64

// ---------------------------------------------------------------------------
// Helpers
// ---------------------------------------------------------------------------
__device__ __forceinline__ u16 f2bf(float f) {
  unsigned u = __float_as_uint(f);
  return (u16)((u + 0x7fffu + ((u >> 16) & 1u)) >> 16);   // RNE
}
__device__ __forceinline__ float bf2f(u16 h) {
  return __uint_as_float(((unsigned)h) << 16);
}

// Load one 16-bit WMMA A/B fragment (16 bf16 per lane). Caller passes
// p = base + row*stride + 8*laneHi (+kk); elements 0..7 = p[0..7],
// elements 8..15 = p[16..23]  (CDNA5 16-bit 16x32 fragment layout).
__device__ __forceinline__ v16bf load_frag(const u16* p) {
  v8bf lo = *reinterpret_cast<const v8bf*>(p);
  v8bf hi = *reinterpret_cast<const v8bf*>(p + 16);
  return __builtin_shufflevector(lo, hi, 0,1,2,3,4,5,6,7,8,9,10,11,12,13,14,15);
}

__device__ __forceinline__ v8f wmma_bf16(v16bf a, v16bf b, v8f c) {
  return __builtin_amdgcn_wmma_f32_16x16x32_bf16(
      false, a, false, b, (short)0, c, false, false);
}

// Async 16-byte copy global -> LDS (ASYNCcnt tracked).
__device__ __forceinline__ void async_b128(void* lds_ptr, const u16* g) {
  unsigned lds_addr = (unsigned)(unsigned long long)lds_ptr; // low 32b = LDS offset
  asm volatile("global_load_async_to_lds_b128 %0, %1, off"
               :: "v"(lds_addr), "v"(g) : "memory");
}
__device__ __forceinline__ void wait_async2() {
  asm volatile("s_wait_asynccnt 2" ::: "memory");
}
__device__ __forceinline__ void wait_async0() {
  asm volatile("s_wait_asynccnt 0" ::: "memory");
}

// ---------------------------------------------------------------------------
// Kernel 1a: f32 -> bf16 elementwise
// ---------------------------------------------------------------------------
__global__ __launch_bounds__(256) void k_cvt_bf16(const float* __restrict__ s,
                                                  u16* __restrict__ d, int n) {
  int g = blockIdx.x * 256 + threadIdx.x;
  if (g < n) d[g] = f2bf(s[g]);
}

// Kernel 1b: W (1024 rows x N cols, f32) -> Wt (N rows x 1024 cols, bf16)
__global__ __launch_bounds__(256) void k_cvt_transpose(const float* __restrict__ s,
                                                       u16* __restrict__ d, int N) {
  int g = blockIdx.x * 256 + threadIdx.x;     // g = n*1024 + k
  if (g >= N * 1024) return;
  int k = g & 1023;
  int n = g >> 10;
  d[g] = f2bf(s[(size_t)k * N + n]);
}

// ---------------------------------------------------------------------------
// Kernel 2/5: GEMM  C[MxN] = A[MxK](bf16,row) * Bt[NxK](bf16,row)^T
// Block: 256 thr (8 waves, 2x4), block tile 128x256, wave tile 64x64.
// 16 WMMA per 16 global b128 loads per 32-deep k-step.
// ---------------------------------------------------------------------------
template <int OUT_F32>
__global__ __launch_bounds__(256) void k_gemm_bt(const u16* __restrict__ A,
                                                 const u16* __restrict__ Bt,
                                                 void* __restrict__ Cout,
                                                 int M, int N, int K) {
  const int tid    = threadIdx.x;
  const int wave   = tid >> 5;
  const int lane   = tid & 31;
  const int laneLo = lane & 15;
  const int laneHi = (lane >> 4) & 1;
  const int wm = wave >> 2;                 // 0..1
  const int wn = wave & 3;                  // 0..3
  const int m0 = blockIdx.y * 128 + wm * 64;
  const int n0 = blockIdx.x * 256 + wn * 64;

  v8f acc[4][4] = {};

  const u16* aRow[4];
  const u16* bRow[4];
#pragma unroll
  for (int i = 0; i < 4; ++i)
    aRow[i] = A + (size_t)(m0 + 16 * i + laneLo) * K + 8 * laneHi;
#pragma unroll
  for (int j = 0; j < 4; ++j)
    bRow[j] = Bt + (size_t)(n0 + 16 * j + laneLo) * K + 8 * laneHi;

  for (int kk = 0; kk < K; kk += 32) {
    v16bf af[4], bfr[4];
#pragma unroll
    for (int i = 0; i < 4; ++i) af[i]  = load_frag(aRow[i] + kk);
#pragma unroll
    for (int j = 0; j < 4; ++j) bfr[j] = load_frag(bRow[j] + kk);
#pragma unroll
    for (int i = 0; i < 4; ++i)
#pragma unroll
      for (int j = 0; j < 4; ++j)
        acc[i][j] = wmma_bf16(af[i], bfr[j], acc[i][j]);
  }

#pragma unroll
  for (int i = 0; i < 4; ++i) {
#pragma unroll
    for (int j = 0; j < 4; ++j) {
      int n = n0 + 16 * j + laneLo;
#pragma unroll
      for (int r = 0; r < 8; ++r) {
        int m = m0 + 16 * i + 8 * laneHi + r;
        if (OUT_F32) ((float*)Cout)[(size_t)m * N + n] = acc[i][j][r];
        else         ((u16*)Cout)[(size_t)m * N + n]   = f2bf(acc[i][j][r]);
      }
    }
  }
}

// ---------------------------------------------------------------------------
// Kernel 3: RoPE + layout.  Q is additionally pre-scaled by 1/sqrt(DH)=0.125
// (exact exponent shift in bf16) so attention needs no score scaling.
//  qkv: bf16 [4096][3072] -> qr,kr: bf16 [b][h][t][64] ; vT: bf16 [b][h][64][t]
// ---------------------------------------------------------------------------
__global__ __launch_bounds__(256) void k_rope_pack(const u16* __restrict__ qkv,
                                                   const float* __restrict__ cosT,
                                                   const float* __restrict__ sinT,
                                                   u16* __restrict__ qr,
                                                   u16* __restrict__ kr,
                                                   u16* __restrict__ vT) {
  int g = blockIdx.x * 256 + threadIdx.x;          // [b:2][t:10][h:4][i:5]
  if (g >= BB * TT * NH * 32) return;
  int i = g & 31;
  int h = (g >> 5) & 15;
  int t = (g >> 9) & 1023;
  int b = g >> 19;
  int d0 = 2 * i, d1 = 2 * i + 1;

  const u16* row = qkv + ((size_t)b * TT + t) * (3 * DM);
  float ce = cosT[t * DH + d0], co = cosT[t * DH + d1];
  float se = sinT[t * DH + d0], so = sinT[t * DH + d1];

  float qe = bf2f(row[h * DH + d0]), qo = bf2f(row[h * DH + d1]);
  float ke = bf2f(row[DM + h * DH + d0]), ko = bf2f(row[DM + h * DH + d1]);

  size_t bhT = ((size_t)(b * NH + h) * TT + t) * DH;
  qr[bhT + d0] = f2bf((qe * ce - qo * se) * 0.125f);
  qr[bhT + d1] = f2bf((qo * co + qe * so) * 0.125f);
  kr[bhT + d0] = f2bf(ke * ce - ko * se);
  kr[bhT + d1] = f2bf(ko * co + ke * so);

  size_t vtb = (size_t)(b * NH + h) * DH;
  vT[(vtb + d0) * TT + t] = row[2 * DM + h * DH + d0];
  vT[(vtb + d1) * TT + t] = row[2 * DM + h * DH + d1];
}

// ---------------------------------------------------------------------------
// Kernel 4: causal flash attention, transposed formulation.
//  grid (B*H, T/128); block 256 thr (8 waves); wave owns 16 q rows.
//  S^T = K(32k x 64d) * Q^T(64d x 16q);  O^T += V^T(64d x 32k) * P^T(32k x 16q)
//  K / V^T tiles double-buffered in LDS via async-to-LDS.
//  Causal compares only on diagonal tiles (wave-uniform predicate).
// ---------------------------------------------------------------------------
__global__ __launch_bounds__(256) void k_attn(const u16* __restrict__ qr,
                                              const u16* __restrict__ kr,
                                              const u16* __restrict__ vT,
                                              u16* __restrict__ ao) {
  __shared__ __align__(16) u16 kbuf[2][32 * 64];   // [k][d]
  __shared__ __align__(16) u16 vbuf[2][64 * 32];   // [d][k]

  const int bh   = blockIdx.x;                 // b*16+h
  const int q0   = blockIdx.y * 128;
  const int tid  = threadIdx.x;
  const int wave = tid >> 5;
  const int lane = tid & 31;
  const int laneLo = lane & 15;
  const int laneHi = (lane >> 4) & 1;
  const int qmin = q0 + wave * 16;             // smallest q in this wave
  const int qrow = qmin + laneLo;              // this lane's q (t index)

  const u16* Kb = kr + (size_t)bh * TT * DH;
  const u16* Vb = vT + (size_t)bh * DH * TT;

  // Q fragments (B-frag of Q^T), resident for the whole kernel.
  const u16* Qp = qr + ((size_t)bh * TT + qrow) * DH + 8 * laneHi;
  v16bf qf0 = load_frag(Qp);        // d 0..31
  v16bf qf1 = load_frag(Qp + 32);   // d 32..63

  v8f  o[4] = {};
  float mrun = -1e30f, lrun = 0.f;

  const int nTiles = (q0 + 128) / 32;

  const int krow = tid >> 3, kc8 = (tid & 7) * 8;   // K tile: 32 x 64
  const int vrow = tid >> 2, vc8 = (tid & 3) * 8;   // V^T tile: 64 x 32
  async_b128(&kbuf[0][krow * 64 + kc8], Kb + (size_t)krow * DH + kc8);
  async_b128(&vbuf[0][vrow * 32 + vc8], Vb + (size_t)vrow * TT + vc8);

  for (int it = 0; it < nTiles; ++it) {
    const int cur = it & 1;
    const int kt  = it * 32;
    if (it + 1 < nTiles) {
      const int nkt = kt + 32, nb = cur ^ 1;
      async_b128(&kbuf[nb][krow * 64 + kc8], Kb + (size_t)(nkt + krow) * DH + kc8);
      async_b128(&vbuf[nb][vrow * 32 + vc8], Vb + (size_t)vrow * TT + nkt + vc8);
      wait_async2();
    } else {
      wait_async0();
    }
    __syncthreads();

    // ---- load all K fragments, then S^T = K * Q^T ----
    v16bf ka[2][2];
#pragma unroll
    for (int j = 0; j < 2; ++j) {
      const u16* kp = &kbuf[cur][(16 * j + laneLo) * 64 + 8 * laneHi];
      ka[j][0] = load_frag(kp);
      ka[j][1] = load_frag(kp + 32);
    }
    v8f st[2] = {};
#pragma unroll
    for (int j = 0; j < 2; ++j) {
      st[j] = wmma_bf16(ka[j][0], qf0, st[j]);
      st[j] = wmma_bf16(ka[j][1], qf1, st[j]);
    }

    // ---- V fragments early: DS latency hides under softmax VALU ----
    v16bf vf[4];
#pragma unroll
    for (int c = 0; c < 4; ++c)
      vf[c] = load_frag(&vbuf[cur][(16 * c + laneLo) * 32 + 8 * laneHi]);

    // ---- masked online softmax (per lane = one q column) ----
    float s[2][8];
    float tmax = -1e30f;
    if (kt + 31 > qmin) {          // diagonal tile for this wave: apply mask
#pragma unroll
      for (int j = 0; j < 2; ++j)
#pragma unroll
        for (int r = 0; r < 8; ++r) {
          int kg = kt + 16 * j + 8 * laneHi + r;
          float v = (kg > qrow) ? -1e30f : st[j][r];
          s[j][r] = v;
          tmax = fmaxf(tmax, v);
        }
    } else {                       // interior tile: no compares
#pragma unroll
      for (int j = 0; j < 2; ++j)
#pragma unroll
        for (int r = 0; r < 8; ++r) {
          s[j][r] = st[j][r];
          tmax = fmaxf(tmax, st[j][r]);
        }
    }
    tmax = fmaxf(tmax, __shfl_xor(tmax, 16, 32));
    float mnew = fmaxf(mrun, tmax);
    float corr = __expf(mrun - mnew);

    float psum = 0.f;
    v16bf pf;
#pragma unroll
    for (int j = 0; j < 2; ++j)
#pragma unroll
      for (int r = 0; r < 8; ++r) {
        float p = __expf(s[j][r] - mnew);
        psum += p;
        pf[j * 8 + r] = (__bf16)p;
      }
    psum += __shfl_xor(psum, 16, 32);
    lrun = lrun * corr + psum;
    mrun = mnew;
#pragma unroll
    for (int c = 0; c < 4; ++c) o[c] = o[c] * corr;

    // ---- O^T += V^T * P^T ----
#pragma unroll
    for (int c = 0; c < 4; ++c) o[c] = wmma_bf16(vf[c], pf, o[c]);

    __syncthreads();   // before buffer `cur` is overwritten next iteration
  }

  // ---- epilogue: normalize, write ao[b][t][h*64+d] as bf16 ----
  const int b = bh >> 4, h = bh & 15;
  const float inv = 1.f / lrun;
  u16* dstRow = ao + ((size_t)(b * TT + qrow)) * DM + h * DH + 8 * laneHi;
#pragma unroll
  for (int c = 0; c < 4; ++c) {
    v8bf ob;
#pragma unroll
    for (int r = 0; r < 8; ++r) ob[r] = (__bf16)(o[c][r] * inv);
    *reinterpret_cast<v8bf*>(dstRow + 16 * c) = ob;
  }
}

// ---------------------------------------------------------------------------
// Host launcher
// ---------------------------------------------------------------------------
extern "C" void kernel_launch(void* const* d_in, const int* in_sizes, int n_in,
                              void* d_out, int out_size, void* d_ws, size_t ws_size,
                              hipStream_t stream) {
  const float* x    = (const float*)d_in[0];   // [4,1024,1024]
  const float* rcos = (const float*)d_in[1];   // [1,1,1024,64]
  const float* rsin = (const float*)d_in[2];
  const float* Wqkv = (const float*)d_in[3];   // [1024,3072]
  const float* Wprj = (const float*)d_in[4];   // [1024,1024]
  float* out = (float*)d_out;

  unsigned char* ws = (unsigned char*)d_ws;
  const size_t M = (size_t)BB * TT;            // 4096
  u16* xb     = (u16*)(ws);                                  // 8  MB
  u16* wqkvT  = (u16*)(ws + (8u  << 20));                    // 6  MB
  u16* wprjT  = (u16*)(ws + (14u << 20));                    // 2  MB
  u16* qkv    = (u16*)(ws + (16u << 20));                    // 24 MB
  u16* qr     = (u16*)(ws + (40u << 20));                    // 8  MB
  u16* kr     = (u16*)(ws + (48u << 20));                    // 8  MB
  u16* vT     = (u16*)(ws + (56u << 20));                    // 8  MB
  u16* ao     = (u16*)(ws + (64u << 20));                    // 8  MB

  // 1) conversions
  {
    int n = (int)(M * DM);
    k_cvt_bf16<<<(n + 255) / 256, 256, 0, stream>>>(x, xb, n);
    k_cvt_transpose<<<(3 * DM * DM + 255) / 256, 256, 0, stream>>>(Wqkv, wqkvT, 3 * DM);
    k_cvt_transpose<<<(DM * DM + 255) / 256, 256, 0, stream>>>(Wprj, wprjT, DM);
  }
  // 2) qkv = xb @ Wqkv            (4096 x 3072, K=1024) -> bf16
  k_gemm_bt<0><<<dim3(3 * DM / 256, M / 128), 256, 0, stream>>>(
      xb, wqkvT, (void*)qkv, (int)M, 3 * DM, DM);
  // 3) rope + repack (Q pre-scaled by 0.125)
  k_rope_pack<<<(BB * TT * NH * 32) / 256, 256, 0, stream>>>(qkv, rcos, rsin, qr, kr, vT);
  // 4) attention
  k_attn<<<dim3(BB * NH, TT / 128), 256, 0, stream>>>(qr, kr, vT, ao);
  // 5) out = ao @ Wproj           (4096 x 1024, K=1024) -> f32
  k_gemm_bt<1><<<dim3(DM / 256, M / 128), 256, 0, stream>>>(
      ao, wprjT, (void*)out, (int)M, DM, DM);
}
